// GlaringDetectorHeadV4_3530463117742
// MI455X (gfx1250) — matrix operand order repaired and checked by reading, not compiled, not run on hardware
//
#include <hip/hip_runtime.h>
#include <hip/hip_bf16.h>
#include <math.h>

// ---------------------------------------------------------------------------
// CDNA5 (gfx1250) WMMA ViT forward.  All matmuls route through
// v_wmma_f32_16x16x32_f16 (f16 in, f32 accumulate).  wave32 everywhere.
// GEMM K-loops are software-pipelined: next K-tile staged global->regs while
// current tile's WMMAs issue from LDS (ping-pong buffers, 1 barrier/step).
// ---------------------------------------------------------------------------

typedef __attribute__((ext_vector_type(16))) _Float16 v16h;
typedef __attribute__((ext_vector_type(8)))  float    v8f;

#define LDSK 40  // LDS row stride (halfs) for 32-wide K tiles; 80B rows dodge bank conflicts

// ---- WMMA fragment loaders (CDNA5 ISA 7.12.2 layouts) ----------------------
// A (16x32 f16): lane L -> row M = L&15; lanes 0-15 hold K {0..7,16..23},
// lanes 16-31 hold K {8..15,24..31}; element pairs packed per VGPR.
static __device__ __forceinline__ v16h frag_a_lds(const _Float16* tile, int mBase) {
  int lane = threadIdx.x & 31;
  const _Float16* p = tile + (size_t)(mBase + (lane & 15)) * LDSK;
  int hs = (lane >> 4) * 8;
  v16h a;
#pragma unroll
  for (int i = 0; i < 8; ++i) a[i] = p[hs + i];
#pragma unroll
  for (int i = 0; i < 8; ++i) a[8 + i] = p[16 + hs + i];
  return a;
}

// B (32x16 f16) stored n-major in LDS ([n][k]): lane L -> col N = L&15,
// lanes 0-15 hold K {0..7,16..23}, lanes 16-31 hold K {8..15,24..31}.
static __device__ __forceinline__ v16h frag_b_lds(const _Float16* tile, int nBase) {
  int lane = threadIdx.x & 31;
  const _Float16* p = tile + (size_t)(nBase + (lane & 15)) * LDSK;
  int hs = (lane >> 4) * 8;
  v16h b;
#pragma unroll
  for (int i = 0; i < 8; ++i) b[i] = p[hs + i];
#pragma unroll
  for (int i = 0; i < 8; ++i) b[8 + i] = p[16 + hs + i];
  return b;
}

static __device__ __forceinline__ float gelu_exact(float v) {
  return 0.5f * v * (1.0f + erff(v * 0.70710678118654752f));
}

// ---------------------------------------------------------------------------
// Generic WMMA GEMM:  C[m][n] = sum_k A[m][k] * B'[n][k]   (i.e. C = A @ W^T)
//   BT=false : B' = Bw, row-major [N][K], ld = ldb
//   BT=true  : B'[n][k] = Bw[k*ldb + n]  (rows contiguous in n; used attn@V)
// Batched via blockIdx.z: z -> (bb = z/Hh, hh = z%Hh), pointer strides per dim.
// EPI: 0 = C+bias ; 1 = gelu(C+bias) ; 2 = C+bias+Res ; 3 = C*scale
// ---------------------------------------------------------------------------
template <int EPI, bool BT>
__global__ void __launch_bounds__(128) gemm_wmma(
    const float* __restrict__ A, int lda, long sAb, long sAh,
    const float* __restrict__ Bw, int ldb, long sBb, long sBh,
    const float* __restrict__ bias,
    const float* __restrict__ Res,
    float* __restrict__ C, int ldc, long sCb, long sCh,
    int M, int N, int K, int Hh, float scale)
{
  __shared__ _Float16 As[2][64 * LDSK];
  __shared__ _Float16 Bs[2][64 * LDSK];

  const int bz = blockIdx.z;
  const int bb = bz / Hh, hh = bz % Hh;
  A  += (size_t)bb * sAb + (size_t)hh * sAh;
  Bw += (size_t)bb * sBb + (size_t)hh * sBh;
  C  += (size_t)bb * sCb + (size_t)hh * sCh;
  if (Res) Res += (size_t)bb * sCb + (size_t)hh * sCh;

  const int m0 = blockIdx.y * 64;
  const int n0 = blockIdx.x * 64;
  const int tid  = threadIdx.x;
  const int warp = tid >> 5;
  const int wy = (warp >> 1) * 32, wx = (warp & 1) * 32;

  v8f acc[2][2] = {};

  // A / non-transposed B staging map: thread -> (tile row, 16-wide k chunk)
  const int lr = tid >> 1;         // 0..63
  const int lc = (tid & 1) * 16;   // 0 or 16
  // transposed-B staging map: thread -> (k row, 16-wide n chunk); n contiguous
  const int kl = tid >> 2;         // 0..31
  const int nc = (tid & 3) * 16;   // 0,16,32,48

  float ra[16], rb[16];

  auto loadA = [&](int k0) {
    int row = m0 + lr;
    if (row < M && (k0 + lc + 16) <= K) {
      const float4* src = reinterpret_cast<const float4*>(A + (size_t)row * lda + k0 + lc);
      __builtin_prefetch((const void*)(src + 8), 0, 1);   // tile after next
#pragma unroll
      for (int q = 0; q < 4; ++q) {
        float4 f = src[q];
        ra[q * 4 + 0] = f.x; ra[q * 4 + 1] = f.y; ra[q * 4 + 2] = f.z; ra[q * 4 + 3] = f.w;
      }
    } else {
#pragma unroll
      for (int q = 0; q < 16; ++q) {
        int k = k0 + lc + q;
        ra[q] = (row < M && k < K) ? A[(size_t)row * lda + k] : 0.f;
      }
    }
  };

  auto loadB = [&](int k0) {
    if (!BT) {
      int n = n0 + lr;
      if (n < N && (k0 + lc + 16) <= K) {
        const float4* src = reinterpret_cast<const float4*>(Bw + (size_t)n * ldb + k0 + lc);
        __builtin_prefetch((const void*)(src + 8), 0, 1);
#pragma unroll
        for (int q = 0; q < 4; ++q) {
          float4 f = src[q];
          rb[q * 4 + 0] = f.x; rb[q * 4 + 1] = f.y; rb[q * 4 + 2] = f.z; rb[q * 4 + 3] = f.w;
        }
      } else {
#pragma unroll
        for (int q = 0; q < 16; ++q) {
          int k = k0 + lc + q;
          rb[q] = (n < N && k < K) ? Bw[(size_t)n * ldb + k] : 0.f;
        }
      }
    } else {
      int k = k0 + kl;
      int nb = n0 + nc;
      if (k < K && nb + 16 <= N) {
        const float4* src = reinterpret_cast<const float4*>(Bw + (size_t)k * ldb + nb);
#pragma unroll
        for (int q = 0; q < 4; ++q) {
          float4 f = src[q];
          rb[q * 4 + 0] = f.x; rb[q * 4 + 1] = f.y; rb[q * 4 + 2] = f.z; rb[q * 4 + 3] = f.w;
        }
      } else {
#pragma unroll
        for (int q = 0; q < 16; ++q) {
          int n = nb + q;
          rb[q] = (k < K && n < N) ? Bw[(size_t)k * ldb + n] : 0.f;
        }
      }
    }
  };

  auto commit = [&](int buf) {
    _Float16* da = &As[buf][lr * LDSK + lc];
#pragma unroll
    for (int q = 0; q < 16; ++q) da[q] = (_Float16)ra[q];
    if (!BT) {
      _Float16* db = &Bs[buf][lr * LDSK + lc];
#pragma unroll
      for (int q = 0; q < 16; ++q) db[q] = (_Float16)rb[q];
    } else {
      _Float16* db = &Bs[buf][0];
#pragma unroll
      for (int q = 0; q < 16; ++q) db[(nc + q) * LDSK + kl] = (_Float16)rb[q];
    }
  };

  // prologue: stage + commit first tile
  loadA(0); loadB(0); commit(0);
  __syncthreads();

  int cur = 0;
  for (int k0 = 0; k0 < K; k0 += 32) {
    const bool more = (k0 + 32) < K;           // uniform across block
    if (more) { loadA(k0 + 32); loadB(k0 + 32); }

    v16h a0 = frag_a_lds(As[cur], wy);
    v16h a1 = frag_a_lds(As[cur], wy + 16);
    v16h b0 = frag_b_lds(Bs[cur], wx);
    v16h b1 = frag_b_lds(Bs[cur], wx + 16);
    acc[0][0] = __builtin_amdgcn_wmma_f32_16x16x32_f16(false, a0, false, b0, (short)0, acc[0][0], false, false);
    acc[0][1] = __builtin_amdgcn_wmma_f32_16x16x32_f16(false, a0, false, b1, (short)0, acc[0][1], false, false);
    acc[1][0] = __builtin_amdgcn_wmma_f32_16x16x32_f16(false, a1, false, b0, (short)0, acc[1][0], false, false);
    acc[1][1] = __builtin_amdgcn_wmma_f32_16x16x32_f16(false, a1, false, b1, (short)0, acc[1][1], false, false);

    if (more) commit(cur ^ 1);
    __syncthreads();
    cur ^= 1;
  }

  // ---- epilogue (C/D layout: VGPR r -> M = r + 8*(lane>>4), N = lane&15) ----
  const int lane = tid & 31;
  const int nl = lane & 15;
  const int mo = (lane >> 4) * 8;
#pragma unroll
  for (int i = 0; i < 2; ++i)
#pragma unroll
    for (int j = 0; j < 2; ++j)
#pragma unroll
      for (int r = 0; r < 8; ++r) {
        int m = m0 + wy + i * 16 + mo + r;
        int n = n0 + wx + j * 16 + nl;
        if (m < M && n < N) {
          float v = acc[i][j][r];
          if constexpr (EPI == 3) {
            v *= scale;
          } else {
            v += bias[n];
            if constexpr (EPI == 1) v = gelu_exact(v);
            if constexpr (EPI == 2) v += Res[(size_t)m * ldc + n];
          }
          C[(size_t)m * ldc + n] = v;
        }
      }
}

// ---------------------------------------------------------------------------
// Implicit-im2col conv patch embed (stride 2, pad 1):  M=B*256, N=E=128,
// K=CIN*9=2304.  Same ping-pong pipeline as gemm_wmma; the im2col gather is
// branch-free: (ci,ky,kx) tracked incrementally (no per-element divides) and
// loads are unconditional from clamped addresses, zero-selected afterwards.
// Epilogue fuses conv bias + pos_emb and scatters to token 1+p.
// ---------------------------------------------------------------------------
__global__ void __launch_bounds__(128) conv_embed_wmma(
    const float* __restrict__ feat,   // [64][256][32][32]
    const float* __restrict__ Wc,     // [128][2304]
    const float* __restrict__ convb,  // [128]
    const float* __restrict__ pos,    // [257][128]
    float* __restrict__ x, int Npad)  // [64][Npad][128]
{
  const int M = 16384, N = 128, K = 2304;
  __shared__ _Float16 As[2][64 * LDSK];
  __shared__ _Float16 Bs[2][64 * LDSK];

  const int m0 = blockIdx.y * 64;
  const int n0 = blockIdx.x * 64;
  const int tid  = threadIdx.x;
  const int warp = tid >> 5;
  const int wy = (warp >> 1) * 32, wx = (warp & 1) * 32;

  v8f acc[2][2] = {};
  const int lr = tid >> 1;
  const int lc = (tid & 1) * 16;

  // per-thread row decode (constant across K)
  const int row = m0 + lr;                 // < M (grid exact)
  const int bimg = row >> 8, p = row & 255;
  const int base_y = (p >> 4) * 2 - 1;     // oy*2 - 1
  const int base_x = (p & 15) * 2 - 1;     // ox*2 - 1
  const float* fb = feat + (size_t)bimg * 256 * 1024;

  float ra[16], rb[16];

  auto loadA = [&](int k0) {
    int k = k0 + lc;
    int ci = k / 9;                        // once per 16-element chunk
    int r9 = k - ci * 9;
    int ky = r9 / 3;
    int kx = r9 - ky * 3;
#pragma unroll
    for (int q = 0; q < 16; ++q) {
      int iy = base_y + ky, ix = base_x + kx;
      bool ok = ((unsigned)iy < 32u) && ((unsigned)ix < 32u);
      int iyc = ok ? iy : 0, ixc = ok ? ix : 0;
      float v = fb[((size_t)ci << 10) + (iyc << 5) + ixc];   // unconditional load
      ra[q] = ok ? v : 0.f;
      // incremental (kx,ky,ci) update -- selects, not divides
      ++kx;
      if (kx == 3) { kx = 0; ++ky; if (ky == 3) { ky = 0; ++ci; } }
    }
  };

  auto loadB = [&](int k0) {
    int n = n0 + lr;
    const float4* src = reinterpret_cast<const float4*>(Wc + (size_t)n * K + k0 + lc);
    __builtin_prefetch((const void*)(src + 8), 0, 1);
#pragma unroll
    for (int q = 0; q < 4; ++q) {
      float4 f = src[q];
      rb[q * 4 + 0] = f.x; rb[q * 4 + 1] = f.y; rb[q * 4 + 2] = f.z; rb[q * 4 + 3] = f.w;
    }
  };

  auto commit = [&](int buf) {
    _Float16* da = &As[buf][lr * LDSK + lc];
    _Float16* db = &Bs[buf][lr * LDSK + lc];
#pragma unroll
    for (int q = 0; q < 16; ++q) da[q] = (_Float16)ra[q];
#pragma unroll
    for (int q = 0; q < 16; ++q) db[q] = (_Float16)rb[q];
  };

  loadA(0); loadB(0); commit(0);
  __syncthreads();

  int cur = 0;
  for (int k0 = 0; k0 < K; k0 += 32) {
    const bool more = (k0 + 32) < K;
    if (more) { loadA(k0 + 32); loadB(k0 + 32); }

    v16h a0 = frag_a_lds(As[cur], wy);
    v16h a1 = frag_a_lds(As[cur], wy + 16);
    v16h b0 = frag_b_lds(Bs[cur], wx);
    v16h b1 = frag_b_lds(Bs[cur], wx + 16);
    acc[0][0] = __builtin_amdgcn_wmma_f32_16x16x32_f16(false, a0, false, b0, (short)0, acc[0][0], false, false);
    acc[0][1] = __builtin_amdgcn_wmma_f32_16x16x32_f16(false, a0, false, b1, (short)0, acc[0][1], false, false);
    acc[1][0] = __builtin_amdgcn_wmma_f32_16x16x32_f16(false, a1, false, b0, (short)0, acc[1][0], false, false);
    acc[1][1] = __builtin_amdgcn_wmma_f32_16x16x32_f16(false, a1, false, b1, (short)0, acc[1][1], false, false);

    if (more) commit(cur ^ 1);
    __syncthreads();
    cur ^= 1;
  }

  const int lane = tid & 31;
  const int nl = lane & 15;
  const int mo = (lane >> 4) * 8;
#pragma unroll
  for (int i = 0; i < 2; ++i)
#pragma unroll
    for (int j = 0; j < 2; ++j)
#pragma unroll
      for (int r = 0; r < 8; ++r) {
        int m = m0 + wy + i * 16 + mo + r;
        int n = n0 + wx + j * 16 + nl;
        if (m < M && n < N) {
          int b = m >> 8, pp = m & 255;
          int tok = 1 + pp;
          float v = acc[i][j][r] + convb[n] + pos[(size_t)tok * 128 + n];
          x[((size_t)b * Npad + tok) * 128 + n] = v;
        }
      }
}

// ---------------------------------------------------------------------------
// Elementwise / reduction kernels
// ---------------------------------------------------------------------------
__global__ void init_x_kernel(float* __restrict__ x, const float* __restrict__ cls,
                              const float* __restrict__ pos, int total, int Npad) {
  int i = blockIdx.x * blockDim.x + threadIdx.x;
  if (i >= total) return;
  int e = i & 127;
  int t = (i >> 7) % Npad;
  float v = 0.f;
  if (t == 0) v = cls[e] + pos[e];   // cls token + pos_emb[0]
  x[i] = v;
}

// LayerNorm over E=128, one row per 128-thread block.
__global__ void __launch_bounds__(128) ln_kernel(const float* __restrict__ x,
                                                 const float* __restrict__ s,
                                                 const float* __restrict__ b,
                                                 float* __restrict__ out, float eps) {
  __shared__ float red[128];
  size_t row = blockIdx.x;
  int e = threadIdx.x;
  float v = x[row * 128 + e];
  red[e] = v; __syncthreads();
  for (int st = 64; st > 0; st >>= 1) { if (e < st) red[e] += red[e + st]; __syncthreads(); }
  float mean = red[0] * (1.f / 128.f);
  __syncthreads();
  float d = v - mean;
  red[e] = d * d; __syncthreads();
  for (int st = 64; st > 0; st >>= 1) { if (e < st) red[e] += red[e + st]; __syncthreads(); }
  float var = red[0] * (1.f / 128.f);
  out[row * 128 + e] = d * rsqrtf(var + eps) * s[e] + b[e];
}

// Masked softmax over one score row (valid cols < Nvalid, pad cols zeroed).
__global__ void __launch_bounds__(128) softmax_kernel(float* __restrict__ S, int Npad, int Nvalid) {
  __shared__ float red[128];
  float* row = S + ((size_t)blockIdx.y * Npad + blockIdx.x) * Npad;
  int t = threadIdx.x;
  float vals[3];
  float mx = -3.402823466e+38f;
#pragma unroll
  for (int i = 0; i < 3; ++i) {
    int m = t + i * 128;
    vals[i] = (m < Nvalid) ? row[m] : -3.402823466e+38f;
    mx = fmaxf(mx, vals[i]);
  }
  red[t] = mx; __syncthreads();
  for (int st = 64; st > 0; st >>= 1) { if (t < st) red[t] = fmaxf(red[t], red[t + st]); __syncthreads(); }
  mx = red[0]; __syncthreads();
  float sum = 0.f;
#pragma unroll
  for (int i = 0; i < 3; ++i) {
    int m = t + i * 128;
    if (m < Nvalid) { vals[i] = expf(vals[i] - mx); sum += vals[i]; }
    else vals[i] = 0.f;
  }
  red[t] = sum; __syncthreads();
  for (int st = 64; st > 0; st >>= 1) { if (t < st) red[t] += red[t + st]; __syncthreads(); }
  float inv = 1.f / red[0];
#pragma unroll
  for (int i = 0; i < 3; ++i) {
    int m = t + i * 128;
    if (m < Npad) row[m] = vals[i] * inv;
  }
}

// Final LN (eps 1e-6) on token 0 + 2-way FC head.  out = [B*2 logits][B*128 feat]
__global__ void __launch_bounds__(128) head_kernel(const float* __restrict__ x,
                                                   const float* __restrict__ fn_s,
                                                   const float* __restrict__ fn_b,
                                                   const float* __restrict__ fc_w,
                                                   const float* __restrict__ fc_b,
                                                   float* __restrict__ out,
                                                   int Npad, int Bsz) {
  __shared__ float red[128];
  int b = blockIdx.x;
  int e = threadIdx.x;
  float v = x[(size_t)b * Npad * 128 + e];
  red[e] = v; __syncthreads();
  for (int st = 64; st > 0; st >>= 1) { if (e < st) red[e] += red[e + st]; __syncthreads(); }
  float mean = red[0] * (1.f / 128.f);
  __syncthreads();
  float d = v - mean;
  red[e] = d * d; __syncthreads();
  for (int st = 64; st > 0; st >>= 1) { if (e < st) red[e] += red[e + st]; __syncthreads(); }
  float var = red[0] * (1.f / 128.f);
  __syncthreads();
  float f = d * rsqrtf(var + 1e-6f) * fn_s[e] + fn_b[e];
  out[Bsz * 2 + (size_t)b * 128 + e] = f;   // out_feat
#pragma unroll
  for (int c = 0; c < 2; ++c) {
    red[e] = f * fc_w[c * 128 + e]; __syncthreads();
    for (int st = 64; st > 0; st >>= 1) { if (e < st) red[e] += red[e + st]; __syncthreads(); }
    if (e == 0) out[b * 2 + c] = red[0] + fc_b[c];
    __syncthreads();
  }
}

// ---------------------------------------------------------------------------
extern "C" void kernel_launch(void* const* d_in, const int* in_sizes, int n_in,
                              void* d_out, int out_size, void* d_ws, size_t ws_size,
                              hipStream_t stream) {
  (void)in_sizes; (void)n_in; (void)out_size; (void)ws_size;

  const float* feat   = (const float*)d_in[0];
  const float* conv_w = (const float*)d_in[1];
  const float* conv_b = (const float*)d_in[2];
  const float* pos    = (const float*)d_in[3];
  const float* cls    = (const float*)d_in[4];
  const float* ln1_s  = (const float*)d_in[5];
  const float* ln1_b  = (const float*)d_in[6];
  const float* qkv_w  = (const float*)d_in[7];
  const float* qkv_b  = (const float*)d_in[8];
  const float* proj_w = (const float*)d_in[9];
  const float* proj_b = (const float*)d_in[10];
  const float* ln2_s  = (const float*)d_in[11];
  const float* ln2_b  = (const float*)d_in[12];
  const float* f1_w   = (const float*)d_in[13];
  const float* f1_b   = (const float*)d_in[14];
  const float* f2_w   = (const float*)d_in[15];
  const float* f2_b   = (const float*)d_in[16];
  const float* fn_s   = (const float*)d_in[17];
  const float* fn_b   = (const float*)d_in[18];
  const float* fc_w   = (const float*)d_in[19];
  const float* fc_b   = (const float*)d_in[20];
  float* out = (float*)d_out;

  const int  Bsz = 64, E = 128, Hh = 4, F = 512, L = 12;
  const int  Npad = 272, Nv = 257;            // 257 tokens padded to 17*16
  const long MR = (long)Bsz * Npad;           // 17408 activation rows

  // workspace carve-up (256B aligned)
  char* wsp = (char*)d_ws;
  auto carve = [&](size_t bytes) { float* p = (float*)wsp; wsp += (bytes + 255) & ~(size_t)255; return p; };
  float* x   = carve((size_t)MR * E * 4);                     // residual stream
  float* h   = carve((size_t)MR * E * 4);                     // LN output
  float* qkv = carve((size_t)MR * 3 * E * 4);                 // packed q|k|v
  float* S   = carve((size_t)Bsz * Hh * Npad * Npad * 4);     // attn scores / probs
  float* o   = carve((size_t)MR * E * 4);                     // attn output
  float* g   = carve((size_t)MR * F * 4);                     // FFN hidden

  // token buffer: zeros + cls/pos on token 0
  {
    int total = (int)(MR * E);
    init_x_kernel<<<(total + 255) / 256, 256, 0, stream>>>(x, cls, pos, total, Npad);
  }
  // conv patch embed -> tokens 1..256 (+bias +pos_emb)
  conv_embed_wmma<<<dim3(2, 256, 1), 128, 0, stream>>>(feat, conv_w, conv_b, pos, x, Npad);

  const float scl = 0.17677669529663687f;     // 1/sqrt(D=32)
  const long sQb = (long)Npad * 3 * E;        // per-batch stride in qkv rows
  const long sSb = (long)Hh * Npad * Npad;    // per-batch stride in S
  const long sSh = (long)Npad * Npad;

  for (int l = 0; l < L; ++l) {
    // h = LN1(x)
    ln_kernel<<<(int)MR, 128, 0, stream>>>(x, ln1_s + (size_t)l * E, ln1_b + (size_t)l * E, h, 1e-5f);
    // qkv = h @ Wqkv^T + b
    gemm_wmma<0, false><<<dim3(6, 272, 1), 128, 0, stream>>>(
        h, E, 0, 0, qkv_w + (size_t)l * 3 * E * E, E, 0, 0,
        qkv_b + (size_t)l * 3 * E, nullptr, qkv, 3 * E, 0, 0,
        (int)MR, 3 * E, E, 1, 1.f);
    // S = (Q @ K^T) * scale   per (b,h)
    gemm_wmma<3, false><<<dim3(5, 5, Bsz * Hh), 128, 0, stream>>>(
        qkv,      3 * E, sQb, 32,
        qkv + E,  3 * E, sQb, 32,
        nullptr, nullptr,
        S, Npad, sSb, sSh,
        Npad, Npad, 32, Hh, scl);
    // masked softmax rows (pad cols -> 0)
    softmax_kernel<<<dim3(Nv, Bsz * Hh, 1), 128, 0, stream>>>(S, Npad, Nv);
    // O = P @ V   (V accessed transposed; contiguous-in-n staging)
    gemm_wmma<3, true><<<dim3(1, 5, Bsz * Hh), 128, 0, stream>>>(
        S, Npad, sSb, sSh,
        qkv + 2 * E, 3 * E, sQb, 32,
        nullptr, nullptr,
        o, E, (long)Npad * E, 32,
        Npad, 32, Npad, Hh, 1.f);
    // x = x + O @ Wp^T + bp   (in place)
    gemm_wmma<2, false><<<dim3(2, 272, 1), 128, 0, stream>>>(
        o, E, 0, 0, proj_w + (size_t)l * E * E, E, 0, 0,
        proj_b + (size_t)l * E, x, x, E, 0, 0,
        (int)MR, E, E, 1, 1.f);
    // h = LN2(x)
    ln_kernel<<<(int)MR, 128, 0, stream>>>(x, ln2_s + (size_t)l * E, ln2_b + (size_t)l * E, h, 1e-5f);
    // g = gelu(h @ W1^T + b1)
    gemm_wmma<1, false><<<dim3(8, 272, 1), 128, 0, stream>>>(
        h, E, 0, 0, f1_w + (size_t)l * F * E, E, 0, 0,
        f1_b + (size_t)l * F, nullptr, g, F, 0, 0,
        (int)MR, F, E, 1, 1.f);
    // x = x + g @ W2^T + b2   (in place)
    gemm_wmma<2, false><<<dim3(2, 272, 1), 128, 0, stream>>>(
        g, F, 0, 0, f2_w + (size_t)l * E * F, F, 0, 0,
        f2_b + (size_t)l * E, x, x, E, 0, 0,
        (int)MR, E, F, 1, 1.f);
  }

  // final LN (eps 1e-6) on cls token + FC head
  head_kernel<<<Bsz, 128, 0, stream>>>(x, fn_s, fn_b, fc_w, fc_b, out, Npad, Bsz);
}
